// HCNN_KNOWN_U_Cell_83322365543010
// MI455X (gfx1250) — matrix operand order repaired and checked by reading, not compiled, not run on hardware
//
#include <hip/hip_runtime.h>

typedef __attribute__((ext_vector_type(16))) _Float16 v16h;
typedef __attribute__((ext_vector_type(8)))  _Float16 v8h;
typedef __attribute__((ext_vector_type(8)))  float    v8f;

#define B_TOTAL   65536
#define NSTATE    256
#define NU        32
#define NY        16
#define KDIM      288          // NY + (NSTATE-NY) + NU
#define ROWS      64           // batch rows per block
#define LDS_STRIDE 296         // f16 elems per LDS row (pad; 592B, 16B aligned)
#define WEIGHT_ELEMS (NSTATE * KDIM)   // 73728

static __device__ __forceinline__ float fast_tanh(float x) {
#if __has_builtin(__builtin_amdgcn_tanhf)
  return __builtin_amdgcn_tanhf(x);    // v_tanh_f32 (gfx1250 TRANS op)
#else
  return tanhf(x);
#endif
}

// ---------------------------------------------------------------------------
// Kernel 0: one-shot (per launch) f32 -> f16 weight conversion into d_ws.
// A_weight is (256 x 288) row-major; same layout in f16 so a WMMA B-fragment
// is 16 contiguous f16 of one row (32B per lane).
// ---------------------------------------------------------------------------
__global__ __launch_bounds__(256)
void hcnn_convert_weight_f16(const float* __restrict__ w,
                             _Float16* __restrict__ w16) {
  int i = blockIdx.x * 256 + threadIdx.x;
  if (i < WEIGHT_ELEMS) w16[i] = (_Float16)w[i];
}

// ---------------------------------------------------------------------------
// Main fused kernel: output diff + tanh(pre) -> LDS f16 -> WMMA GEMM.
// Block: 256 threads = 8 waves; covers 64 batch rows x 256 output cols.
// Wave w: row-group rg = w&3 (16 rows), col-group cg = w>>2 (128 cols).
// ---------------------------------------------------------------------------
__global__ __launch_bounds__(256)
void hcnn_cell_kernel(const float* __restrict__ state,
                      const float* __restrict__ known,
                      const float* __restrict__ obs,
                      const _Float16* __restrict__ w16,
                      float* __restrict__ new_state,
                      float* __restrict__ outp) {
  __shared__ _Float16 lds[ROWS * LDS_STRIDE];   // 37,888 bytes

  const int tid  = threadIdx.x;
  const int lane = tid & 31;
  const int wave = tid >> 5;
  const long b0  = (long)blockIdx.x * ROWS;

  // Warm L2/WGP$ with the f16 weight while phase 1 streams the state.
  for (long off = (long)tid * 64; off < WEIGHT_ELEMS; off += 256 * 64)
    __builtin_prefetch(w16 + off, 0, 1);

  // ---------------- Phase 1: rstate = tanh(pre) into LDS (f16) -------------
  // pre[:, 0:16]   = observation           (expectation - output == obs)
  // pre[:, 16:256] = state[:, 16:256]
  // pre[:, 256:288]= known_features
  // Also: output[b, c] = state[b, c] - obs[b, c] for c < 16.
  // Streamed-once data uses non-temporal policy to keep the weight cached.
  for (int rr = wave; rr < ROWS; rr += 8) {
    const long b = b0 + rr;
    const float* srow = state + b * NSTATE;
#pragma unroll
    for (int j = 0; j < KDIM / 32; ++j) {
      const int c = lane + j * 32;
      float x;
      if (c < NY) {
        const float ob = __builtin_nontemporal_load(obs + b * NY + c);
        const float st = __builtin_nontemporal_load(srow + c);
        __builtin_nontemporal_store(st - ob, outp + b * NY + c);
        x = ob;
      } else if (c < NSTATE) {
        x = __builtin_nontemporal_load(srow + c);
      } else {
        x = __builtin_nontemporal_load(known + b * NU + (c - NSTATE));
      }
      lds[rr * LDS_STRIDE + c] = (_Float16)fast_tanh(x);
    }
  }
  __syncthreads();

  // ---------------- Phase 2: new_state = rstate @ A_weight^T ---------------
  const int rg = wave & 3;          // row-group: rows m0 .. m0+15
  const int cg = wave >> 2;         // col-group: cols n0 .. n0+127
  const int m0 = rg * 16;
  const int n0 = cg * 128;
  const int lhalf = lane >> 4;      // 0: lanes 0-15, 1: lanes 16-31
  const int l16   = lane & 15;

  v8f acc[8] = {};                  // 8 N-tiles of 16x16 f32

  // A-fragment: lanes 0-15 hold K = kb+{0..15},
  // lanes 16-31 hold K = kb+{16..31}; M = m0 + l16, from LDS.
  const _Float16* arow = &lds[(m0 + l16) * LDS_STRIDE + lhalf * 16];
  // B-fragment: lane holds N = n0 + nt*16 + l16, 16 contiguous K starting at
  // ks*32 + lhalf*16 -> one 32B read of a weight row.
  const _Float16* wbase = w16 + (long)(n0 + l16) * KDIM + lhalf * 16;

#define LOAD_A(ks)                                                            \
  __builtin_shufflevector(*(const v8h*)(arow + (ks) * 32),                    \
                          *(const v8h*)(arow + (ks) * 32 + 8),                \
                          0, 1, 2, 3, 4, 5, 6, 7, 8, 9, 10, 11, 12, 13, 14, 15)
#define LOAD_B(ks, nt) (*(const v16h*)(wbase + (long)(nt) * (16 * KDIM) + (ks) * 32))

  // Flattened 72-step pipeline: t = ks*8 + nt. B-fragments rotate through 3
  // distinct buffers (depth-2 pipeline) so loads stay in flight across WMMAs
  // instead of forcing s_wait_loadcnt 0 before every matrix op.
  v16h aF = LOAD_A(0);
  v16h aN = aF;
  v16h bbuf[3];
  bbuf[0] = LOAD_B(0, 0);
  bbuf[1] = LOAD_B(0, 1);

#pragma unroll
  for (int t = 0; t < 72; ++t) {
    const int ks = t >> 3;
    const int nt = t & 7;
    if (nt == 0 && ks < 8) aN = LOAD_A(ks + 1);   // prefetch next A from LDS
    const int tp = t + 2;                          // keep 2 B-frags in flight
    if (tp < 72) bbuf[tp % 3] = LOAD_B(tp >> 3, tp & 7);
    acc[nt] = __builtin_amdgcn_wmma_f32_16x16x32_f16(
        /*neg_a=*/false, aF, /*neg_b=*/false, bbuf[t % 3],
        /*c_mod=*/(short)0, acc[nt],
        /*reuse_a=*/false, /*reuse_b=*/false);
    if (nt == 7) aF = aN;
  }
#undef LOAD_A
#undef LOAD_B

  // C/D layout: VGPR r holds M = r + 8*lhalf, lane l16 holds N.
  // Results are streamed once -> non-temporal stores.
#pragma unroll
  for (int nt = 0; nt < 8; ++nt) {
    const long n = n0 + nt * 16 + l16;
#pragma unroll
    for (int r = 0; r < 8; ++r) {
      const long row = b0 + m0 + r + 8 * lhalf;
      __builtin_nontemporal_store(acc[nt][r], new_state + row * NSTATE + n);
    }
  }
}

// ---------------------------------------------------------------------------
extern "C" void kernel_launch(void* const* d_in, const int* in_sizes, int n_in,
                              void* d_out, int out_size, void* d_ws, size_t ws_size,
                              hipStream_t stream) {
  const float* state = (const float*)d_in[0];   // (B, 256)
  const float* known = (const float*)d_in[1];   // (B, 32)
  const float* obs   = (const float*)d_in[2];   // (B, 16)
  const float* aw    = (const float*)d_in[3];   // (256, 288)

  float* new_state = (float*)d_out;                               // B*256
  float* outp      = (float*)d_out + (size_t)B_TOTAL * NSTATE;    // B*16

  _Float16* w16 = (_Float16*)d_ws;  // needs 147,456 bytes of scratch

  hcnn_convert_weight_f16<<<(WEIGHT_ELEMS + 255) / 256, 256, 0, stream>>>(aw, w16);
  hcnn_cell_kernel<<<B_TOTAL / ROWS, 256, 0, stream>>>(
      state, known, obs, w16, new_state, outp);
}
